// MainModel_69329362092111
// MI455X (gfx1250) — matrix operand order repaired and checked by reading, not compiled, not run on hardware
//
#include <hip/hip_runtime.h>
#include <hip/hip_bf16.h>

#define N_NODES 500000
#define N_EDGES 8000000
#define BATCH   100000

typedef __attribute__((ext_vector_type(16))) __bf16 v16bf;
typedef __attribute__((ext_vector_type(8)))  float  v8f;

// ---------------------------------------------------------------------------
// agg_w = segment_sum(weight, dst)   (the +1.0 is folded in at divide time)
// ---------------------------------------------------------------------------
__global__ void aggw_kernel(const int* __restrict__ dst, const float* __restrict__ w,
                            float* __restrict__ aggw, int n_edges) {
  int tid = blockIdx.x * blockDim.x + threadIdx.x;
  int stride = gridDim.x * blockDim.x;
  for (int e = tid; e < n_edges; e += stride) {
    __builtin_prefetch(&dst[e + stride], 0, 1);
    atomicAdd(&aggw[dst[e]], w[e]);
  }
}

// ---------------------------------------------------------------------------
// agg[dst] += z[src] * w   — one edge per DIM-lane group (DIM = 32 or 8) so
// the DIM-float gather and atomic scatter are coalesced across lanes.
// dst_limit lets layer-2 skip rows beyond the batch.
// ---------------------------------------------------------------------------
template<int DIM>
__global__ void scatter_kernel(const float* __restrict__ z, const int* __restrict__ src,
                               const int* __restrict__ dst, const float* __restrict__ w,
                               float* __restrict__ agg, int n_edges, int dst_limit) {
  int tid = blockIdx.x * blockDim.x + threadIdx.x;
  int e = tid / DIM;
  int c = tid % DIM;
  int stride = (gridDim.x * blockDim.x) / DIM;
  for (; e < n_edges; e += stride) {
    __builtin_prefetch(&src[e + stride], 0, 1);
    int d = dst[e];
    if (d < dst_limit) {
      float val = z[(size_t)src[e] * DIM + c] * w[e];
      atomicAdd(&agg[(size_t)d * DIM + c], val);
    }
  }
}

// ---------------------------------------------------------------------------
// Y = relu( concat(X1, X2/(aggw+1)) @ W + bias ), row-major, via
// v_wmma_f32_16x16x32_bf16. One 16-row tile per wave iteration; B fragments
// (weights, <=8KB) loaded once per wave; K zero-padded to 32, N to 16.
// ---------------------------------------------------------------------------
template<int IN1, int IN2, int OUT>
__global__ void wmma_mlp(const float* __restrict__ X1, const float* __restrict__ X2,
                         const float* __restrict__ aggw,
                         const float* __restrict__ W, const float* __restrict__ bias,
                         float* __restrict__ Y, int nrows) {
  constexpr int K  = IN1 + IN2;
  constexpr int KB = (K + 31) / 32;   // 32-wide K steps
  constexpr int NB = (OUT + 15) / 16; // 16-wide N blocks
  const int lane  = threadIdx.x & 31;
  const int wave  = (blockIdx.x * blockDim.x + threadIdx.x) >> 5;
  const int nwav  = (gridDim.x * blockDim.x) >> 5;
  const int half  = lane >> 4;   // 0: lanes 0-15, 1: lanes 16-31
  const int m     = lane & 15;

  // B fragments: element j of lane covers K = kb*32 + j + 16*half, N = m
  v16bf bfrag[NB][KB];
  #pragma unroll
  for (int nb = 0; nb < NB; ++nb) {
    int n = nb * 16 + m;
    #pragma unroll
    for (int kb = 0; kb < KB; ++kb) {
      v16bf bf{};
      #pragma unroll
      for (int j = 0; j < 16; ++j) {
        int k = kb * 32 + j + 16 * half;
        float v = (k < K && n < OUT) ? W[(size_t)k * OUT + n] : 0.0f;
        bf[j] = (__bf16)v;
      }
      bfrag[nb][kb] = bf;
    }
  }
  float bias_v[NB];
  #pragma unroll
  for (int nb = 0; nb < NB; ++nb) {
    int n = nb * 16 + m;
    bias_v[nb] = (n < OUT) ? bias[n] : 0.0f;
  }

  int ntiles = nrows >> 4;  // nrows is a multiple of 16 (500000, 100000)
  for (int t = wave; t < ntiles; t += nwav) {
    int row = t * 16 + m;
    float invw = 1.0f;
    if constexpr (IN2 > 0) invw = 1.0f / (aggw[row] + 1.0f);

    // A fragments: element j covers K = kb*32 + j + 16*half
    v16bf afrag[KB];
    #pragma unroll
    for (int kb = 0; kb < KB; ++kb) {
      v16bf af{};
      #pragma unroll
      for (int j = 0; j < 16; ++j) {
        int k = kb * 32 + j + 16 * half;
        float v = 0.0f;
        if (k < IN1) v = X1[(size_t)row * IN1 + k];
        else if constexpr (IN2 > 0) {
          if (k < K) v = X2[(size_t)row * IN2 + (k - IN1)] * invw;
        }
        af[j] = (__bf16)v;
      }
      afrag[kb] = af;
    }

    #pragma unroll
    for (int nb = 0; nb < NB; ++nb) {
      v8f c;
      #pragma unroll
      for (int i = 0; i < 8; ++i) c[i] = bias_v[nb];
      #pragma unroll
      for (int kb = 0; kb < KB; ++kb)
        c = __builtin_amdgcn_wmma_f32_16x16x32_bf16(
              false, afrag[kb], false, bfrag[nb][kb], (short)0, c, false, false);
      int n = nb * 16 + m;
      if (n < OUT) {
        #pragma unroll
        for (int i = 0; i < 8; ++i) {
          int r = t * 16 + half * 8 + i;   // C/D layout: M = half*8 + vgpr
          float v = c[i];
          Y[(size_t)r * OUT + n] = v > 0.0f ? v : 0.0f;
        }
      }
    }
  }
}

// ---------------------------------------------------------------------------
// Final loss: one thread per group of 5 rows (ctr, pos, 3 neg), 40-dim
// embeddings = concat(pos32, temp8). LDS block-reduce then global atomic.
// ---------------------------------------------------------------------------
__global__ void loss_kernel(const float* __restrict__ ep, const float* __restrict__ et,
                            const float* __restrict__ cent, const float* __restrict__ cent_t,
                            float* __restrict__ out, int b) {
  int g = blockIdx.x * blockDim.x + threadIdx.x;
  float contrib = 0.0f;
  if (g < b) {
    float z0[40];
    int r0 = g * 5;
    #pragma unroll
    for (int d = 0; d < 32; ++d) z0[d] = ep[(size_t)r0 * 32 + d];
    #pragma unroll
    for (int d = 0; d < 8; ++d) z0[32 + d] = et[(size_t)r0 * 8 + d];

    auto softmaxC = [&](const float* v, float* qout) {
      float l[5];
      #pragma unroll
      for (int c = 0; c < 5; ++c) {
        float s = 0.0f;
        #pragma unroll
        for (int d = 0; d < 32; ++d) s += v[d] * cent[c * 32 + d];
        #pragma unroll
        for (int d = 0; d < 8; ++d) s += v[32 + d] * cent_t[c * 8 + d];
        l[c] = s;
      }
      float mx = l[0];
      #pragma unroll
      for (int c = 1; c < 5; ++c) mx = fmaxf(mx, l[c]);
      float se = 0.0f;
      #pragma unroll
      for (int c = 0; c < 5; ++c) { l[c] = __expf(l[c] - mx); se += l[c]; }
      float inv = 1.0f / se;
      #pragma unroll
      for (int c = 0; c < 5; ++c) qout[c] = l[c] * inv;
    };

    float q[5][5];
    float dotr[5];
    softmaxC(z0, q[0]);
    for (int i = 1; i < 5; ++i) {
      float zi[40];
      int r = r0 + i;
      #pragma unroll
      for (int d = 0; d < 32; ++d) zi[d] = ep[(size_t)r * 32 + d];
      #pragma unroll
      for (int d = 0; d < 8; ++d) zi[32 + d] = et[(size_t)r * 8 + d];
      float s = 0.0f;
      #pragma unroll
      for (int d = 0; d < 40; ++d) s += z0[d] * zi[d];
      dotr[i] = s;
      softmaxC(zi, q[i]);
    }
    float pd = dotr[1];
    float nd = fmaxf(dotr[2], fmaxf(dotr[3], dotr[4]));
    float mm = fmaxf(nd - pd + 1.0f, 0.0f);

    float qq[5];
    #pragma unroll
    for (int i = 1; i < 5; ++i) {
      float s = 0.0f;
      #pragma unroll
      for (int c = 0; c < 5; ++c) s += q[0][c] * q[i][c];
      qq[i] = s;
    }
    float cmm = fmaxf(fmaxf(qq[2], fmaxf(qq[3], qq[4])) - qq[1] + 1.0f, 0.0f);

    float ne = 0.0f;
    #pragma unroll
    for (int i = 0; i < 5; ++i)
      #pragma unroll
      for (int c = 0; c < 5; ++c) ne += q[i][c] * __logf(q[i][c] + 1e-20f);

    contrib = (mm + cmm) / (float)b + 0.01f * ne / (5.0f * (float)b);
  }
  __shared__ float acc;
  if (threadIdx.x == 0) acc = 0.0f;
  __syncthreads();
  atomicAdd(&acc, contrib);
  __syncthreads();
  if (threadIdx.x == 0) atomicAdd(out, acc);
}

// ---------------------------------------------------------------------------
extern "C" void kernel_launch(void* const* d_in, const int* in_sizes, int n_in,
                              void* d_out, int out_size, void* d_ws, size_t ws_size,
                              hipStream_t stream) {
  const float* x      = (const float*)d_in[0];
  const float* wgt    = (const float*)d_in[1];
  const float* e_w1l  = (const float*)d_in[2];
  const float* e_b1l  = (const float*)d_in[3];
  const float* e_w1r  = (const float*)d_in[4];
  const float* e_b1r  = (const float*)d_in[5];
  const float* e_w2l  = (const float*)d_in[6];
  const float* e_b2l  = (const float*)d_in[7];
  const float* e_w2r  = (const float*)d_in[8];
  const float* e_b2r  = (const float*)d_in[9];
  const float* t_w1l  = (const float*)d_in[10];
  const float* t_b1l  = (const float*)d_in[11];
  const float* t_w1r  = (const float*)d_in[12];
  const float* t_b1r  = (const float*)d_in[13];
  const float* t_w2l  = (const float*)d_in[14];
  const float* t_b2l  = (const float*)d_in[15];
  const float* t_w2r  = (const float*)d_in[16];
  const float* t_b2r  = (const float*)d_in[17];
  const float* cent   = (const float*)d_in[18];
  const float* cent_t = (const float*)d_in[19];
  const int*   ei     = (const int*)d_in[20];
  const int* src = ei;
  const int* dst = ei + N_EDGES;
  float* out = (float*)d_out;

  const size_t N = N_NODES;
  float* ws   = (float*)d_ws;
  float* aggw = ws;                         // N
  float* bufA = aggw + N;                   // N*32
  float* bufB = bufA + N * 32;              // N*32
  float* h1p  = bufB + N * 32;              // N*32
  float* h2p  = h1p  + N * 32;              // BATCH*32
  float* h2t  = h2p  + (size_t)BATCH * 32;  // BATCH*8
  // temporal overlays inside bufA/bufB (positional module is finished first)
  float* zt   = bufA;
  float* h1t  = bufA + N * 8;
  float* z2t  = bufA + N * 16;
  float* aggt = bufB;

  dim3 blk(256);
  const int gemmN = 2048, gemmB = 512, scat = 4096, edg = 2048;

  // edge-degree weights, computed once, reused by all 4 layers
  hipMemsetAsync(aggw, 0, N * sizeof(float), stream);
  aggw_kernel<<<edg, blk, 0, stream>>>(dst, wgt, aggw, N_EDGES);

  // ---- positional module (dim 32) ----
  wmma_mlp<32, 0, 32><<<gemmN, blk, 0, stream>>>(x, nullptr, nullptr, e_w1l, e_b1l, bufA, N_NODES);
  hipMemsetAsync(bufB, 0, N * 32 * sizeof(float), stream);
  scatter_kernel<32><<<scat, blk, 0, stream>>>(bufA, src, dst, wgt, bufB, N_EDGES, N_NODES);
  wmma_mlp<32, 32, 32><<<gemmN, blk, 0, stream>>>(x, bufB, aggw, e_w1r, e_b1r, h1p, N_NODES);
  wmma_mlp<32, 0, 32><<<gemmN, blk, 0, stream>>>(h1p, nullptr, nullptr, e_w2l, e_b2l, bufA, N_NODES);
  hipMemsetAsync(bufB, 0, (size_t)BATCH * 32 * sizeof(float), stream);
  scatter_kernel<32><<<scat, blk, 0, stream>>>(bufA, src, dst, wgt, bufB, N_EDGES, BATCH);
  wmma_mlp<32, 32, 32><<<gemmB, blk, 0, stream>>>(h1p, bufB, aggw, e_w2r, e_b2r, h2p, BATCH);

  // ---- temporal module (dim 8) ----
  wmma_mlp<32, 0, 8><<<gemmN, blk, 0, stream>>>(x, nullptr, nullptr, t_w1l, t_b1l, zt, N_NODES);
  hipMemsetAsync(aggt, 0, N * 8 * sizeof(float), stream);
  scatter_kernel<8><<<scat, blk, 0, stream>>>(zt, src, dst, wgt, aggt, N_EDGES, N_NODES);
  wmma_mlp<32, 8, 8><<<gemmN, blk, 0, stream>>>(x, aggt, aggw, t_w1r, t_b1r, h1t, N_NODES);
  wmma_mlp<8, 0, 8><<<gemmN, blk, 0, stream>>>(h1t, nullptr, nullptr, t_w2l, t_b2l, z2t, N_NODES);
  hipMemsetAsync(aggt, 0, (size_t)BATCH * 8 * sizeof(float), stream);
  scatter_kernel<8><<<scat, blk, 0, stream>>>(z2t, src, dst, wgt, aggt, N_EDGES, BATCH);
  wmma_mlp<8, 8, 8><<<gemmB, blk, 0, stream>>>(h1t, aggt, aggw, t_w2r, t_b2r, h2t, BATCH);

  // ---- loss ----
  hipMemsetAsync(out, 0, sizeof(float), stream);
  int b = BATCH / 5;
  loss_kernel<<<(b + 255) / 256, blk, 0, stream>>>(h2p, h2t, cent, cent_t, out, b);
}
